// CharacteristicTransport2D_22299470201665
// MI455X (gfx1250) — matrix operand order repaired and checked by reading, not compiled, not run on hardware
//
#include <hip/hip_runtime.h>
#include <hip/hip_bf16.h>

typedef __bf16 bf16_t;
typedef __attribute__((ext_vector_type(16))) __bf16 v16bf;
typedef __attribute__((ext_vector_type(8)))  float  v8f;

// Problem constants (from reference): B=8, N=4096 (64x64), D=384, G=4, C=96
#define BATCH   8
#define NTOK    4096
#define DIM     384
#define MROWS   (BATCH * NTOK)      // 32768
#define NCAT    2688                // 768(state) + 384(s1) + 384(ga) + 384(gb) + 768(inj)
#define GRID_H  64
#define GRID_W  64

__device__ __forceinline__ bf16_t f2bf(float f) {
  union { float f; unsigned u; } in; in.f = f;
  unsigned u = in.u;
  unsigned r = (u + 0x7FFFu + ((u >> 16) & 1u)) >> 16;  // round-to-nearest-even
  union { unsigned short s; bf16_t h; } out; out.s = (unsigned short)r;
  return out.h;
}

__device__ __forceinline__ float sigmoidf_(float z) {
  return 1.0f / (1.0f + __expf(-z));
}

// ---------------------------------------------------------------------------
// Per-wave 64(M) x 32(N) GEMM tile, K in steps of 32, bf16 in / f32 accumulate.
// A: row-major MxK.
// Bp: fragment-major packed:  Bp[((k/32)*N + n)*32 + s] = B[(k/32)*32 + s][n]
// so each lane's 16-element B fragment is 32 contiguous bytes.
// A and B both use contiguous k order (k = 16*half + s) per lane half (wave32).
// ---------------------------------------------------------------------------
__device__ __forceinline__ void wave_gemm_tile(
    const bf16_t* __restrict__ A, int lda,
    const bf16_t* __restrict__ Bp, int N,
    int K, int m0, int n0, v8f acc[4][2])
{
  const int lane = threadIdx.x & 31;
  const int half = lane >> 4;
  const int r    = lane & 15;
  for (int k = 0; k < K; k += 32) {
    v16bf afrag[4];
#pragma unroll
    for (int mi = 0; mi < 4; ++mi) {
      const bf16_t* ap = A + (size_t)(m0 + mi * 16 + r) * lda + k + half * 16;
#pragma unroll
      for (int j = 0; j < 4; ++j) {
        afrag[mi][2 * j]     = ap[2 * j];          // K = 2j   (+16*half)
        afrag[mi][2 * j + 1] = ap[2 * j + 1];      // K = 2j+1
        afrag[mi][8 + 2 * j] = ap[8 + 2 * j];      // K = 8+2j
        afrag[mi][9 + 2 * j] = ap[8 + 2 * j + 1];  // K = 8+2j+1
      }
    }
    v16bf bfrag[2];
#pragma unroll
    for (int ni = 0; ni < 2; ++ni) {
      const bf16_t* bp = Bp + (((size_t)(k >> 5) * N + (n0 + ni * 16 + r)) << 5)
                            + (half << 4);
#pragma unroll
      for (int t = 0; t < 16; ++t) bfrag[ni][t] = bp[t];   // 32 contiguous bytes
    }
#pragma unroll
    for (int mi = 0; mi < 4; ++mi)
#pragma unroll
      for (int ni = 0; ni < 2; ++ni)
        acc[mi][ni] = __builtin_amdgcn_wmma_f32_16x16x32_bf16(
            false, afrag[mi], false, bfrag[ni], (short)0, acc[mi][ni],
            false, false);
  }
}

// ---------------------------------------------------------------------------
// Pack / convert kernels
// ---------------------------------------------------------------------------
__global__ void k_f32_to_bf16(const float* __restrict__ src,
                              bf16_t* __restrict__ dst, long n) {
  long i = (long)blockIdx.x * blockDim.x + threadIdx.x;
  if (i < n) dst[i] = f2bf(src[i]);
}

// Fragment-major pack of the concatenated projection weights (K=384, N=2688).
// dst[(kb*NCAT + n)*32 + s] = W[kb*32 + s][n] with region dispatch over n.
__global__ void k_pack_wcat_frag(const float* __restrict__ Wstate,
                                 const float* __restrict__ W1,
                                 const float* __restrict__ Wga,
                                 const float* __restrict__ Wgb,
                                 const float* __restrict__ Winj,
                                 bf16_t* __restrict__ Wcat) {
  int i = blockIdx.x * blockDim.x + threadIdx.x;
  if (i >= DIM * NCAT) return;
  int s  = i & 31;
  int rr = i >> 5;
  int n  = rr % NCAT;
  int kb = rr / NCAT;
  int k  = kb * 32 + s;
  float val;
  if      (n <  768) val = Wstate[(size_t)k * 768 + n];
  else if (n < 1152) val = W1 [(size_t)k * 384 + (n -  768)];
  else if (n < 1536) val = Wga[(size_t)k * 384 + (n - 1152)];
  else if (n < 1920) val = Wgb[(size_t)k * 384 + (n - 1536)];
  else               val = Winj[(size_t)k * 768 + (n - 1920)];
  Wcat[i] = f2bf(val);
}

// Generic fragment-major pack of a row-major KxN f32 matrix.
__global__ void k_pack_bfrag(const float* __restrict__ src,
                             bf16_t* __restrict__ dst, int K, int N) {
  long i = (long)blockIdx.x * blockDim.x + threadIdx.x;
  if (i >= (long)K * N) return;
  int s  = (int)(i & 31);
  long rr = i >> 5;
  int n  = (int)(rr % N);
  int kb = (int)(rr / N);
  dst[i] = f2bf(src[(size_t)(kb * 32 + s) * N + n]);
}

__global__ void k_pack_bias(const float* __restrict__ b1,
                            const float* __restrict__ bga,
                            const float* __restrict__ bgb,
                            float* __restrict__ biascat) {
  int c = blockIdx.x * blockDim.x + threadIdx.x;
  if (c >= NCAT) return;
  float v = 0.0f;
  if      (c >=  768 && c < 1152) v = b1 [c -  768];
  else if (c >= 1152 && c < 1536) v = bga[c - 1152];
  else if (c >= 1536 && c < 1920) v = bgb[c - 1536];
  biascat[c] = v;
}

// ---------------------------------------------------------------------------
// Fused projection GEMM: (32768 x 384) x (384 x 2688), per-region epilogue.
// Grid: (MROWS/128, NCAT/128) = (256, 21), block 256 (8 waves, 2x4 tiling).
// ---------------------------------------------------------------------------
__global__ __launch_bounds__(256) void k_gemm_proj(
    const bf16_t* __restrict__ Xb, const bf16_t* __restrict__ Wcat,
    const float* __restrict__ biascat,
    float* __restrict__ u,  float* __restrict__ v,
    float* __restrict__ s1,
    float* __restrict__ ga, float* __restrict__ gb,
    float* __restrict__ xu, float* __restrict__ xv)
{
  const int wid = threadIdx.x >> 5;
  const int wm = wid >> 2, wn = wid & 3;
  const int m0 = blockIdx.x * 128 + wm * 64;
  const int n0 = blockIdx.y * 128 + wn * 32;

  v8f acc[4][2];
#pragma unroll
  for (int mi = 0; mi < 4; ++mi)
#pragma unroll
    for (int ni = 0; ni < 2; ++ni)
#pragma unroll
      for (int i = 0; i < 8; ++i) acc[mi][ni][i] = 0.0f;

  wave_gemm_tile(Xb, DIM, Wcat, NCAT, DIM, m0, n0, acc);

  const int lane = threadIdx.x & 31;
  const int half = lane >> 4, r = lane & 15;
#pragma unroll
  for (int mi = 0; mi < 4; ++mi) {
#pragma unroll
    for (int ni = 0; ni < 2; ++ni) {
      const int ncol = n0 + ni * 16 + r;
      const float bias = biascat[ncol];
#pragma unroll
      for (int i = 0; i < 8; ++i) {
        const int m = m0 + mi * 16 + i + 8 * half;
        const int b = m >> 12, n = m & (NTOK - 1);
        const float z = acc[mi][ni][i] + bias;
        if (ncol < 768) {                 // uv = x @ W_state -> grid layout
          float* dst = (ncol < 384) ? u : v;
          int c = (ncol < 384) ? ncol : ncol - 384;
          dst[((size_t)b * DIM + c) * NTOK + n] = z;
        } else if (ncol < 1152) {         // silu(x@W1+b1) -> token-major s1
          s1[(size_t)m * DIM + (ncol - 768)] = z * sigmoidf_(z);
        } else if (ncol < 1536) {         // gate_a
          ga[((size_t)b * DIM + (ncol - 1152)) * NTOK + n] = sigmoidf_(z);
        } else if (ncol < 1920) {         // gate_b
          gb[((size_t)b * DIM + (ncol - 1536)) * NTOK + n] = sigmoidf_(z);
        } else {                          // inj -> xu / xv grids
          int c = ncol - 1920;
          float* dst = (c < 384) ? xu : xv;
          if (c >= 384) c -= 384;
          dst[((size_t)b * DIM + c) * NTOK + n] = z;
        }
      }
    }
  }
}

// ---------------------------------------------------------------------------
// psi[b,g,n] = sum_c s1[m,c] * W_stream2[c,g] + b_stream2[g]
// ---------------------------------------------------------------------------
__global__ void k_psi(const float* __restrict__ s1, const float* __restrict__ W2,
                      const float* __restrict__ b2, float* __restrict__ psi) {
  int m = blockIdx.x * blockDim.x + threadIdx.x;
  if (m >= MROWS) return;
  int b = m >> 12, n = m & (NTOK - 1);
  float a0 = 0, a1 = 0, a2 = 0, a3 = 0;
  const float* sp = s1 + (size_t)m * DIM;
  for (int c = 0; c < DIM; ++c) {
    float s = sp[c];
    a0 += s * W2[c * 4 + 0]; a1 += s * W2[c * 4 + 1];
    a2 += s * W2[c * 4 + 2]; a3 += s * W2[c * 4 + 3];
  }
  size_t base = (size_t)b * 4 * NTOK + n;
  psi[base + 0 * NTOK] = a0 + b2[0];
  psi[base + 1 * NTOK] = a1 + b2[1];
  psi[base + 2 * NTOK] = a2 + b2[2];
  psi[base + 3 * NTOK] = a3 + b2[3];
}

// ---------------------------------------------------------------------------
// routing: stencil velocities + softmax over 5 directions
// ---------------------------------------------------------------------------
__global__ void k_routing(const float* __restrict__ psi,
                          const float* __restrict__ self_bias,
                          float* __restrict__ rout) {
  int idx = blockIdx.x * blockDim.x + threadIdx.x;   // over B*G*NTOK
  if (idx >= BATCH * 4 * NTOK) return;
  int n = idx & (NTOK - 1), bg = idx >> 12;
  int g = bg & 3;
  int y = n >> 6, xw = n & 63;
  const float* p = psi + (size_t)bg * NTOK;
  float up = (y  >  0) ? p[n - GRID_W] : 0.0f;
  float dn = (y  < 63) ? p[n + GRID_W] : 0.0f;
  float lf = (xw >  0) ? p[n - 1]      : 0.0f;
  float rt = (xw < 63) ? p[n + 1]      : 0.0f;
  float vx =  0.5f * (dn - up);
  float vy = -0.5f * (rt - lf);
  const float inv_tau = 2.0f;                         // 1 / TAU, TAU = 0.5
  float l[5] = { self_bias[g], -vx * inv_tau, vx * inv_tau,
                 -vy * inv_tau, vy * inv_tau };
  float mx = l[0];
#pragma unroll
  for (int j = 1; j < 5; ++j) mx = fmaxf(mx, l[j]);
  float e[5], s = 0.0f;
#pragma unroll
  for (int j = 0; j < 5; ++j) { e[j] = __expf(l[j] - mx); s += e[j]; }
  float inv = 1.0f / s;
  size_t base = (size_t)bg * 5 * NTOK + n;
#pragma unroll
  for (int j = 0; j < 5; ++j) rout[base + (size_t)j * NTOK] = e[j] * inv;
}

// ---------------------------------------------------------------------------
// Transport: all k steps fused in LDS. One block = (b, g, 8 channels).
// LDS: routing 5 planes + u/v double buffers = 9 * 16KB = 144KB (CDNA5 WGP).
// Gates / injection held in registers (16 cells per thread).
// ---------------------------------------------------------------------------
__global__ __launch_bounds__(256) void k_transport(
    float* __restrict__ u, float* __restrict__ v,
    const float* __restrict__ ga, const float* __restrict__ gb,
    const float* __restrict__ xu, const float* __restrict__ xv,
    const float* __restrict__ rout, const int* __restrict__ k_steps)
{
  extern __shared__ float sm[];
  float* srt = sm;                    // 5 * 4096 routing planes
  float* buf = sm + 5 * NTOK;         // 4 * 4096 state buffers
  const int tid   = threadIdx.x;
  const int chunk = blockIdx.x % 12;            // 12 chunks of 8 channels
  const int g     = (blockIdx.x / 12) & 3;
  const int b     = blockIdx.x / 48;
  const int steps = k_steps[0];

  const float* rsrc = rout + ((size_t)(b * 4 + g)) * 5 * NTOK;
  for (int i = tid; i < 5 * NTOK; i += 256) srt[i] = rsrc[i];

  for (int cc = 0; cc < 8; ++cc) {
    const int d = g * 96 + chunk * 8 + cc;
    const size_t base = ((size_t)b * DIM + d) * NTOK;
    float rga[16], rgb[16], rxu[16], rxv[16];
    float* uc = buf;             float* un = buf + NTOK;
    float* vc = buf + 2 * NTOK;  float* vn = buf + 3 * NTOK;
    __syncthreads();   // routing loaded / prior channel fully consumed
#pragma unroll
    for (int j = 0; j < 16; ++j) {
      int n = tid + j * 256;
      rga[j] = ga[base + n]; rgb[j] = gb[base + n];
      rxu[j] = xu[base + n]; rxv[j] = xv[base + n];
      uc[n] = u[base + n];   vc[n] = v[base + n];
    }
    __syncthreads();
    for (int s = 0; s < steps; ++s) {
#pragma unroll
      for (int j = 0; j < 16; ++j) {
        int n = tid + j * 256;
        int y = n >> 6, xw = n & 63;
        float r0 = srt[n],            r1 = srt[NTOK + n],
              r2 = srt[2 * NTOK + n], r3 = srt[3 * NTOK + n],
              r4 = srt[4 * NTOK + n];
        float uu = uc[n] * r0, vv = vc[n] * r0;
        if (y  >  0) { uu += uc[n - GRID_W] * r1; vv += vc[n - GRID_W] * r1; }
        if (y  < 63) { uu += uc[n + GRID_W] * r2; vv += vc[n + GRID_W] * r2; }
        if (xw >  0) { uu += uc[n - 1] * r3;      vv += vc[n - 1] * r3; }
        if (xw < 63) { uu += uc[n + 1] * r4;      vv += vc[n + 1] * r4; }
        un[n] = rga[j] * uu + rgb[j] * rxu[j];
        vn[n] = rga[j] * vv + rgb[j] * rxv[j];
      }
      __syncthreads();
      float* t = uc; uc = un; un = t;
      t = vc; vc = vn; vn = t;
    }
#pragma unroll
    for (int j = 0; j < 16; ++j) {
      int n = tid + j * 256;
      u[base + n] = uc[n]; v[base + n] = vc[n];
    }
  }
}

// ---------------------------------------------------------------------------
// Build bf16 token-major combined = concat(u_out, v_out): (32768 x 768)
// ---------------------------------------------------------------------------
__global__ void k_combine(const float* __restrict__ u,
                          const float* __restrict__ v,
                          bf16_t* __restrict__ comb) {
  size_t i = (size_t)blockIdx.x * blockDim.x + threadIdx.x;
  if (i >= (size_t)BATCH * 768 * NTOK) return;
  int n = (int)(i & (NTOK - 1));
  int c = (int)((i >> 12) % 768);
  int b = (int)(i / ((size_t)768 * NTOK));
  float val = (c < 384) ? u[((size_t)b * DIM + c) * NTOK + n]
                        : v[((size_t)b * DIM + (c - 384)) * NTOK + n];
  comb[((size_t)(b * NTOK + n)) * 768 + c] = f2bf(val);
}

// ---------------------------------------------------------------------------
// Output GEMM: (32768 x 768) x (768 x 384) + x * D_skip
// ---------------------------------------------------------------------------
__global__ __launch_bounds__(256) void k_gemm_out(
    const bf16_t* __restrict__ Cb, const bf16_t* __restrict__ Wob,
    const float* __restrict__ x, const float* __restrict__ Dskip,
    float* __restrict__ out)
{
  const int wid = threadIdx.x >> 5;
  const int wm = wid >> 2, wn = wid & 3;
  const int m0 = blockIdx.x * 128 + wm * 64;
  const int n0 = blockIdx.y * 128 + wn * 32;

  v8f acc[4][2];
#pragma unroll
  for (int mi = 0; mi < 4; ++mi)
#pragma unroll
    for (int ni = 0; ni < 2; ++ni)
#pragma unroll
      for (int i = 0; i < 8; ++i) acc[mi][ni][i] = 0.0f;

  wave_gemm_tile(Cb, 768, Wob, DIM, 768, m0, n0, acc);

  const int lane = threadIdx.x & 31;
  const int half = lane >> 4, r = lane & 15;
#pragma unroll
  for (int mi = 0; mi < 4; ++mi) {
#pragma unroll
    for (int ni = 0; ni < 2; ++ni) {
      const int j = n0 + ni * 16 + r;
      const float dsk = Dskip[j];
#pragma unroll
      for (int i = 0; i < 8; ++i) {
        const int m = m0 + mi * 16 + i + 8 * half;
        out[(size_t)m * DIM + j] = acc[mi][ni][i] + x[(size_t)m * DIM + j] * dsk;
      }
    }
  }
}

// ---------------------------------------------------------------------------
extern "C" void kernel_launch(void* const* d_in, const int* in_sizes, int n_in,
                              void* d_out, int out_size, void* d_ws, size_t ws_size,
                              hipStream_t stream) {
  const float* x         = (const float*)d_in[0];
  const float* W_stream1 = (const float*)d_in[1];
  const float* b_stream1 = (const float*)d_in[2];
  const float* W_stream2 = (const float*)d_in[3];
  const float* b_stream2 = (const float*)d_in[4];
  const float* self_bias = (const float*)d_in[5];
  const float* W_ga      = (const float*)d_in[6];
  const float* b_ga      = (const float*)d_in[7];
  const float* W_gb      = (const float*)d_in[8];
  const float* b_gb      = (const float*)d_in[9];
  const float* W_inj     = (const float*)d_in[10];
  const float* W_state   = (const float*)d_in[11];
  const float* W_out     = (const float*)d_in[12];
  const float* D_skip    = (const float*)d_in[13];
  const int*   k_steps   = (const int*)d_in[14];
  float* out = (float*)d_out;

  // Workspace layout
  char* ws = (char*)d_ws;
  size_t off = 0;
  auto take = [&](size_t bytes) -> void* {
    off = (off + 255) & ~(size_t)255;
    void* p = ws + off;
    off += bytes;
    return p;
  };
  const size_t gridBytes = (size_t)BATCH * DIM * NTOK * sizeof(float); // 50.3MB
  bf16_t* xb     = (bf16_t*)take((size_t)MROWS * DIM * sizeof(bf16_t));
  bf16_t* wcat   = (bf16_t*)take((size_t)DIM * NCAT * sizeof(bf16_t));
  bf16_t* woutb  = (bf16_t*)take((size_t)768 * DIM * sizeof(bf16_t));
  float*  biascat= (float*) take((size_t)NCAT * sizeof(float));
  float*  u      = (float*) take(gridBytes);
  float*  v      = (float*) take(gridBytes);
  float*  ga     = (float*) take(gridBytes);
  float*  gb     = (float*) take(gridBytes);
  float*  xu     = (float*) take(gridBytes);
  float*  xv     = (float*) take(gridBytes);
  // s1 (f32, 50.3MB) is fully consumed by k_psi before k_combine writes
  // the bf16 combined matrix (same 50.3MB) -> alias the same region.
  void*   s1c    = take((size_t)MROWS * DIM * sizeof(float));
  float*  s1     = (float*)s1c;
  bf16_t* comb   = (bf16_t*)s1c;
  float*  psi    = (float*) take((size_t)BATCH * 4 * NTOK * sizeof(float));
  float*  rout   = (float*) take((size_t)BATCH * 4 * 5 * NTOK * sizeof(float));
  (void)ws_size; (void)in_sizes; (void)n_in; (void)out_size;

  const int smemTransport = 9 * NTOK * (int)sizeof(float);  // 147456 B
  hipFuncSetAttribute((const void*)k_transport,
                      hipFuncAttributeMaxDynamicSharedMemorySize, smemTransport);

  // 1) pack x (row-major bf16) and weights (fragment-major bf16)
  {
    long n = (long)MROWS * DIM;
    k_f32_to_bf16<<<(unsigned)((n + 255) / 256), 256, 0, stream>>>(x, xb, n);
  }
  k_pack_wcat_frag<<<(DIM * NCAT + 255) / 256, 256, 0, stream>>>(
      W_state, W_stream1, W_ga, W_gb, W_inj, wcat);
  k_pack_bias<<<(NCAT + 255) / 256, 256, 0, stream>>>(b_stream1, b_ga, b_gb, biascat);
  {
    long n = (long)768 * DIM;
    k_pack_bfrag<<<(unsigned)((n + 255) / 256), 256, 0, stream>>>(
        W_out, woutb, 768, DIM);
  }

  // 2) fused projection GEMM (WMMA bf16, f32 acc)
  k_gemm_proj<<<dim3(MROWS / 128, NCAT / 128), 256, 0, stream>>>(
      xb, wcat, biascat, u, v, s1, ga, gb, xu, xv);

  // 3) psi + routing
  k_psi<<<MROWS / 256, 256, 0, stream>>>(s1, W_stream2, b_stream2, psi);
  k_routing<<<(BATCH * 4 * NTOK) / 256, 256, 0, stream>>>(psi, self_bias, rout);

  // 4) fused k-step transport in LDS (144KB per workgroup)
  k_transport<<<BATCH * 4 * 12, 256, smemTransport, stream>>>(
      u, v, ga, gb, xu, xv, rout, k_steps);

  // 5) combined bf16 (aliased over s1) + output GEMM with skip epilogue
  {
    size_t n = (size_t)BATCH * 768 * NTOK;
    k_combine<<<(unsigned)((n + 255) / 256), 256, 0, stream>>>(u, v, comb);
  }
  k_gemm_out<<<dim3(MROWS / 128, DIM / 128), 256, 0, stream>>>(
      comb, woutb, x, D_skip, out);
}
